// CorrBlock_13486197309758
// MI455X (gfx1250) — matrix operand order repaired and checked by reading, not compiled, not run on hardware
//
#include <hip/hip_runtime.h>

// ---- problem constants ----
#define Bq   2
#define Cc   256
#define Hh   72
#define Ww   72
#define Nn   8
#define HWc  (Hh * Ww)        // 5184
#define BHWc (Bq * HWc)       // 10368
#define RAD  4
#define KK   9
#define LVLS 4
#define CT   (LVLS * KK * KK) // 324

typedef _Float16 half_t;
typedef __attribute__((ext_vector_type(16))) _Float16 v16h;
typedef __attribute__((ext_vector_type(8)))  _Float16 v8h;
typedef __attribute__((ext_vector_type(8)))  float    v8f;
typedef __attribute__((ext_vector_type(4)))  float    v4f;
typedef int v4i_t __attribute__((vector_size(4 * sizeof(int))));

// pyramid level offsets (in halves) within the pyramid buffer
#define PSZ0 ((size_t)BHWc * 5184)
#define PSZ1 ((size_t)BHWc * 1296)
#define PSZ2 ((size_t)BHWc * 324)
#define POFF0 ((size_t)0)
#define POFF1 (POFF0 + PSZ0)
#define POFF2 (POFF1 + PSZ1)
#define POFF3 (POFF2 + PSZ2)

// CDNA5 async global->LDS staging (ASYNCcnt-tracked); guarded so either
// toolchain flavor still compiles (falls back to plain LDS staging).
#if __has_builtin(__builtin_amdgcn_global_load_async_to_lds_b128) && \
    __has_builtin(__builtin_amdgcn_s_wait_asynccnt)
#define USE_ASYNC_LDS 1
#else
#define USE_ASYNC_LDS 0
#endif

// ---------------------------------------------------------------------------
// 1) L2-normalize over C and pack to f16 [pixel][C] panels (GEMM layout).
//    Lane = pixel -> fmap reads are coalesced (fmap is [C][HW], contiguous hw).
// ---------------------------------------------------------------------------
__global__ __launch_bounds__(128)
void norm_pack_kernel(const float* __restrict__ f1, const float* __restrict__ f2,
                      half_t* __restrict__ pa, half_t* __restrict__ pb) {
    const float* src = (blockIdx.y == 0) ? f1 : f2;
    half_t*      dst = (blockIdx.y == 0) ? pa : pb;
    const int p = blockIdx.x * 128 + threadIdx.x;       // 0 .. 10367
    if (p >= BHWc) return;
    const int b  = p / HWc;
    const int hw = p - b * HWc;
    const float* base = src + (size_t)b * Cc * HWc + hw;

    float ss = 0.f;
#pragma unroll 8
    for (int c = 0; c < Cc; ++c) {
        float v = base[(size_t)c * HWc];
        ss += v * v;
    }
    const float inv = 1.0f / fmaxf(__builtin_sqrtf(ss), 1e-12f);

    half_t* o = dst + (size_t)p * Cc;
#pragma unroll 8
    for (int c = 0; c < Cc; ++c)
        o[c] = (half_t)(base[(size_t)c * HWc] * inv);
}

// ---------------------------------------------------------------------------
// 2) All-pairs correlation GEMM: LDS-staged WMMA pipeline.
//    Block = 128 threads = 4 waves (2x2), block tile 64x64, wave tile 32x32.
//    Two K=128 phases: async-stage A/B panels into LDS (ASYNCcnt), then feed
//    v_wmma_f32_16x16x32_f16 from ds_load. Row pitch 136 halves (272 B = 68
//    dwords, 68 mod 64 = 4) -> the 16 fragment rows hit 16 distinct banks.
// ---------------------------------------------------------------------------
#define KCH   128                 // K halves staged per phase
#define PITCH 136                 // padded LDS row pitch (halves)

__global__ __launch_bounds__(128)
void corr_gemm_kernel(const half_t* __restrict__ pa, const half_t* __restrict__ pb,
                      half_t* __restrict__ corr0) {
    __shared__ half_t sA[64 * PITCH];
    __shared__ half_t sB[64 * PITCH];

    const int tid  = threadIdx.x;
    const int lane = tid & 31;
    const int wave = tid >> 5;                  // 0..3
    const int wr = wave >> 1, wc = wave & 1;
    const int b = blockIdx.z;
    const int brow0 = blockIdx.y * 64;          // block row base (A panel)
    const int bcol0 = blockIdx.x * 64;          // block col base (B panel)
    const int lr = lane & 15;                   // row/col-in-fragment
    const int hs = lane >> 4;                   // K half-select

    const half_t* Ag = pa + ((size_t)b * HWc + brow0) * Cc;
    const half_t* Bg = pb + ((size_t)b * HWc + bcol0) * Cc;

    // per-lane LDS fragment row bases (ISA 7.12.2 layouts)
    const int arow[2] = { (wr * 32 + lr) * PITCH, (wr * 32 + 16 + lr) * PITCH };
    const int bcol[2] = { (wc * 32 + lr) * PITCH, (wc * 32 + 16 + lr) * PITCH };

    v8f acc[2][2] = {};

    for (int phase = 0; phase < 2; ++phase) {
        const int kbase = phase * KCH;
        // ---- stage A/B (64 rows x 128 halves each) into LDS ----
#pragma unroll
        for (int i = 0; i < 8; ++i) {
            const int chunk = tid + i * 128;     // 0..1023 (16B units)
            const int row = chunk >> 4;          // 0..63
            const int col = (chunk & 15) * 8;    // half offset within row
            const half_t* gA = Ag + (size_t)row * Cc + kbase + col;
            const half_t* gB = Bg + (size_t)row * Cc + kbase + col;
            half_t* lA = &sA[row * PITCH + col];
            half_t* lB = &sB[row * PITCH + col];
#if USE_ASYNC_LDS
            __builtin_amdgcn_global_load_async_to_lds_b128((v4i_t*)gA, (v4i_t*)lA, 0, 0);
            __builtin_amdgcn_global_load_async_to_lds_b128((v4i_t*)gB, (v4i_t*)lB, 0, 0);
#else
            *(v4f*)lA = *(const v4f*)gA;
            *(v4f*)lB = *(const v4f*)gB;
#endif
        }
#if USE_ASYNC_LDS
        __builtin_amdgcn_s_wait_asynccnt(0);
#endif
        __syncthreads();

        // ---- consume: 4 K-steps of 32, 4 wmma each ----
#pragma unroll
        for (int kk = 0; kk < KCH; kk += 32) {
            const int abase = kk + hs * 8;       // A lane: K {ab..ab+7, ab+16..ab+23}
            const int bbase = kk + hs * 8;       // B lane: K {bb..bb+7, bb+16..bb+23}

            v16h afrag[2], bfrag[2];
#pragma unroll
            for (int f = 0; f < 2; ++f) {
                v8h alo = *(const v8h*)(&sA[arow[f] + abase]);
                v8h ahi = *(const v8h*)(&sA[arow[f] + abase + 16]);
                v8h blo = *(const v8h*)(&sB[bcol[f] + bbase]);
                v8h bhi = *(const v8h*)(&sB[bcol[f] + bbase + 16]);
#pragma unroll
                for (int i = 0; i < 8; ++i) {
                    afrag[f][i] = alo[i]; afrag[f][i + 8] = ahi[i];
                    bfrag[f][i] = blo[i]; bfrag[f][i + 8] = bhi[i];
                }
            }
#pragma unroll
            for (int fi = 0; fi < 2; ++fi)
#pragma unroll
                for (int fj = 0; fj < 2; ++fj)
                    acc[fi][fj] = __builtin_amdgcn_wmma_f32_16x16x32_f16(
                        false, afrag[fi], false, bfrag[fj],
                        (short)0, acc[fi][fj], false, false);
        }
        __syncthreads();    // WAR: finish reads before next-phase staging
    }

    half_t* Cb = corr0 + (size_t)b * HWc * HWc;
#pragma unroll
    for (int fi = 0; fi < 2; ++fi)
#pragma unroll
        for (int fj = 0; fj < 2; ++fj) {
            const int m0 = brow0 + wr * 32 + fi * 16;
            const int n0 = bcol0 + wc * 32 + fj * 16;
#pragma unroll
            for (int v = 0; v < 8; ++v) {
                const int r = m0 + v + 8 * hs;  // C/D layout: VGPR v -> M = v (+8 upper half)
                Cb[(size_t)r * HWc + (n0 + lr)] = (half_t)acc[fi][fj][v];
            }
        }
}

// ---------------------------------------------------------------------------
// 3) 2x2 average pool, level l -> l+1 (f16 in/out, f32 math).
// ---------------------------------------------------------------------------
__global__ void pool2_kernel(const half_t* __restrict__ in, half_t* __restrict__ out,
                             int hin, long n_out) {
    long idx = (long)blockIdx.x * blockDim.x + threadIdx.x;
    if (idx >= n_out) return;
    const int hout  = hin >> 1;
    const int plane = hout * hout;
    const long q   = idx / plane;
    const int rem  = (int)(idx - q * plane);
    const int y = rem / hout, x = rem - (rem / hout) * hout;
    const half_t* ip = in + q * (long)(hin * hin);
    const int y2 = 2 * y, x2 = 2 * x;
    float s = (float)ip[y2 * hin + x2]       + (float)ip[y2 * hin + x2 + 1]
            + (float)ip[(y2 + 1) * hin + x2] + (float)ip[(y2 + 1) * hin + x2 + 1];
    out[idx] = (half_t)(0.25f * s);
}

// ---------------------------------------------------------------------------
// 4) Bilinear 9x9 window sampling over the pyramid. One thread per output
//    element, in the reference's pre-reshape flat layout.
// ---------------------------------------------------------------------------
__global__ void sample_kernel(const float* __restrict__ coords,
                              const half_t* __restrict__ pyr,
                              float* __restrict__ out, long total) {
    long idx = (long)blockIdx.x * blockDim.x + threadIdx.x;
    if (idx >= total) return;

    const int c = (int)(idx % CT);
    long t = idx / CT;
    const int w = (int)(t % Ww); t /= Ww;
    const int h = (int)(t % Hh); t /= Hh;
    const int n = (int)(t % Nn); t /= Nn;
    const int b = (int)t;

    const int level = c / (KK * KK);
    const int rem   = c - level * (KK * KK);
    const int ix = rem / KK;       // x-offset index (outer per reference)
    const int iy = rem - ix * KK;  // y-offset index (inner)

    const size_t cb = (size_t)((b * Nn + n) * 2) * HWc + (size_t)h * Ww + w;
    const float cx = coords[cb];
    const float cy = coords[cb + HWc];

    const float scale = (float)(1 << level);
    const float xs = cx / scale + (float)(ix - RAD);
    const float ys = cy / scale + (float)(iy - RAD);

    const int hl = Hh >> level;
    size_t off;
    switch (level) {
        case 0:  off = POFF0; break;
        case 1:  off = POFF1; break;
        case 2:  off = POFF2; break;
        default: off = POFF3; break;
    }
    const long q = ((long)b * Hh + h) * Ww + w;
    const half_t* plane = pyr + off + (size_t)q * hl * hl;

    const float x0f = floorf(xs), y0f = floorf(ys);
    const int x0 = (int)x0f, y0 = (int)y0f;
    const float wx1 = xs - x0f, wy1 = ys - y0f;

    float acc = 0.f;
#pragma unroll
    for (int dy = 0; dy < 2; ++dy) {
#pragma unroll
        for (int dx = 0; dx < 2; ++dx) {
            const int xi = x0 + dx, yi = y0 + dy;
            if (xi >= 0 && xi < hl && yi >= 0 && yi < hl) {
                const float wgt = (dx ? wx1 : 1.f - wx1) * (dy ? wy1 : 1.f - wy1);
                acc += (float)plane[yi * hl + xi] * wgt;
            }
        }
    }
    out[idx] = acc;
}

// ---------------------------------------------------------------------------
extern "C" void kernel_launch(void* const* d_in, const int* in_sizes, int n_in,
                              void* d_out, int out_size, void* d_ws, size_t ws_size,
                              hipStream_t stream) {
    const float* fmap1  = (const float*)d_in[0];
    const float* fmap2  = (const float*)d_in[1];
    const float* coords = (const float*)d_in[2];
    float* out = (float*)d_out;

    // workspace layout (halves): packedA | packedB | pyramid(4 levels)
    half_t* pa  = (half_t*)d_ws;
    half_t* pb  = pa + (size_t)BHWc * Cc;
    half_t* pyr = pb + (size_t)BHWc * Cc;

    // 1) normalize + pack both maps
    norm_pack_kernel<<<dim3(BHWc / 128, 2), 128, 0, stream>>>(fmap1, fmap2, pa, pb);

    // 2) all-pairs correlation (level-0 corr volume, f16)
    corr_gemm_kernel<<<dim3(HWc / 64, HWc / 64, Bq), 128, 0, stream>>>(pa, pb, pyr + POFF0);

    // 3) pyramid pooling
    {
        long n1 = (long)BHWc * 36 * 36;
        pool2_kernel<<<(unsigned)((n1 + 255) / 256), 256, 0, stream>>>(pyr + POFF0, pyr + POFF1, 72, n1);
        long n2 = (long)BHWc * 18 * 18;
        pool2_kernel<<<(unsigned)((n2 + 255) / 256), 256, 0, stream>>>(pyr + POFF1, pyr + POFF2, 36, n2);
        long n3 = (long)BHWc * 9 * 9;
        pool2_kernel<<<(unsigned)((n3 + 255) / 256), 256, 0, stream>>>(pyr + POFF2, pyr + POFF3, 18, n3);
    }

    // 4) bilinear window sampling -> output
    const long total = (long)Bq * Nn * Hh * Ww * CT;   // 26,873,856
    sample_kernel<<<(unsigned)((total + 255) / 256), 256, 0, stream>>>(coords, pyr, out, total);
}
